// AlexNetOWT_BN_56203942036133
// MI455X (gfx1250) — 8-bit matrix operand order repaired and checked by reading, compiled, not run on hardware
//
#include <hip/hip_runtime.h>
#include <hip/hip_bf16.h>
#include <stdint.h>

typedef __attribute__((ext_vector_type(8))) int v8i;

#define BN_EPS 1e-5f

__device__ __forceinline__ v8i load_a_frag(const int8_t* p) {
  const int4* q = (const int4*)p;
  int4 q0 = q[0], q1 = q[2];
  v8i a;
  a[0] = q0.x; a[1] = q0.y; a[2] = q0.z; a[3] = q0.w;
  a[4] = q1.x; a[5] = q1.y; a[6] = q1.z; a[7] = q1.w;
  return a;
}
__device__ __forceinline__ v8i load_b_frag(const int8_t* p) {
  const int4* q = (const int4*)p;
  int4 q0 = q[0], q1 = q[2];
  v8i b;
  b[0] = q0.x; b[1] = q0.y; b[2] = q0.z; b[3] = q0.w;
  b[4] = q1.x; b[5] = q1.y; b[6] = q1.z; b[7] = q1.w;
  return b;
}

#define WMMA_IU8(A, B, Cacc) \
  __builtin_amdgcn_wmma_i32_16x16x64_iu8(true, (A), true, (B), (Cacc), false, false)

__global__ __launch_bounds__(256, 1)
void gemm_iu8_wmma(const int8_t* __restrict__ A,
                   const int8_t* __restrict__ B,
                   int* __restrict__ C,
                   int M, int N, int K, int Mt32)
{
  const int wave = (int)((blockIdx.x * blockDim.x + threadIdx.x) >> 5);
  const int Nt64 = N >> 6;
  if (wave >= Mt32 * Nt64) return;
  const int lane = threadIdx.x & 31;
  const int r    = lane & 15;
  const int half = lane >> 4;
  const int tm = (wave % Mt32) << 5;
  const int tn = (wave / Mt32) << 6;

  const int8_t* a0p = A + (size_t)(tm + r)      * K + half * 16;
  const int8_t* a1p = A + (size_t)(tm + 16 + r) * K + half * 16;
  const int8_t* b0p = B + (size_t)(tn +  0 + r) * K + half * 16;
  const int8_t* b1p = B + (size_t)(tn + 16 + r) * K + half * 16;
  const int8_t* b2p = B + (size_t)(tn + 32 + r) * K + half * 16;
  const int8_t* b3p = B + (size_t)(tn + 48 + r) * K + half * 16;

  v8i acc00 = (v8i){0,0,0,0,0,0,0,0};
  v8i acc01 = acc00, acc02 = acc00, acc03 = acc00;
  v8i acc10 = acc00, acc11 = acc00, acc12 = acc00, acc13 = acc00;

  for (int kit = K >> 6; kit > 0; --kit) {
    v8i a0 = load_a_frag(a0p);
    v8i a1 = load_a_frag(a1p);
    v8i b0 = load_b_frag(b0p);
    v8i b1 = load_b_frag(b1p);
    v8i b2 = load_b_frag(b2p);
    v8i b3 = load_b_frag(b3p);
    __builtin_prefetch(a0p + 64, 0, 3);
    __builtin_prefetch(a1p + 64, 0, 3);
    __builtin_prefetch(b0p + 64, 0, 3);
    __builtin_prefetch(b1p + 64, 0, 3);
    __builtin_prefetch(b2p + 64, 0, 3);
    __builtin_prefetch(b3p + 64, 0, 3);
    acc00 = WMMA_IU8(a0, b0, acc00);
    acc01 = WMMA_IU8(a0, b1, acc01);
    acc02 = WMMA_IU8(a0, b2, acc02);
    acc03 = WMMA_IU8(a0, b3, acc03);
    acc10 = WMMA_IU8(a1, b0, acc10);
    acc11 = WMMA_IU8(a1, b1, acc11);
    acc12 = WMMA_IU8(a1, b2, acc12);
    acc13 = WMMA_IU8(a1, b3, acc13);
    a0p += 64; a1p += 64;
    b0p += 64; b1p += 64; b2p += 64; b3p += 64;
  }

#pragma unroll
  for (int v = 0; v < 8; ++v) {
    const int m0 = tm + half * 8 + v;
    const int m1 = m0 + 16;
    if (m0 < M) {
      int* row = C + (size_t)m0 * N + tn + r;
      row[0]  = acc00[v];
      row[16] = acc01[v];
      row[32] = acc02[v];
      row[48] = acc03[v];
    }
    if (m1 < M) {
      int* row = C + (size_t)m1 * N + tn + r;
      row[0]  = acc10[v];
      row[16] = acc11[v];
      row[32] = acc12[v];
      row[48] = acc13[v];
    }
  }
}

__global__ void pack_sign(const float* __restrict__ w, int8_t* __restrict__ out,
                          long long nreal, long long ntotal)
{
  long long i = (long long)blockIdx.x * blockDim.x + threadIdx.x;
  if (i >= ntotal) return;
  out[i] = (i < nreal) ? (int8_t)(w[i] >= 0.f ? 1 : -1) : (int8_t)0;
}

__global__ void conv0_direct(const float* __restrict__ x, const int8_t* __restrict__ ws,
                             const float* __restrict__ bias, float* __restrict__ y)
{
  const long long total = 128LL * 64 * 55 * 55;
  long long idx = (long long)blockIdx.x * blockDim.x + threadIdx.x;
  if (idx >= total) return;
  int ow = (int)(idx % 55); long long t = idx / 55;
  int oh = (int)(t % 55); t /= 55;
  int oc = (int)(t % 64); int b = (int)(t / 64);
  float acc = bias[oc];
  const int ih0 = oh * 4 - 2, iw0 = ow * 4 - 2;
  for (int c = 0; c < 3; ++c) {
    for (int kh = 0; kh < 11; ++kh) {
      int ih = ih0 + kh;
      if ((unsigned)ih >= 224u) continue;
      const float*  xr = x  + (((size_t)b * 3 + c) * 224 + ih) * 224;
      const int8_t* wr = ws + (((size_t)oc * 3 + c) * 11 + kh) * 11;
      for (int kw = 0; kw < 11; ++kw) {
        int iw = iw0 + kw;
        if ((unsigned)iw >= 224u) continue;
        acc += xr[iw] * (float)wr[kw];
      }
    }
  }
  y[idx] = acc;
}

__global__ void maxpool_f32(const float* __restrict__ in, float* __restrict__ out,
                            int BC, int H, int W, int OH, int OW)
{
  long long total = (long long)BC * OH * OW;
  long long idx = (long long)blockIdx.x * blockDim.x + threadIdx.x;
  if (idx >= total) return;
  int ow = (int)(idx % OW); long long t = idx / OW;
  int oh = (int)(t % OH);  int bc = (int)(t / OH);
  const float* p = in + (size_t)bc * H * W + (size_t)(oh * 2) * W + ow * 2;
  float m = -3.402823466e38f;
  for (int dh = 0; dh < 3; ++dh)
    for (int dw = 0; dw < 3; ++dw)
      m = fmaxf(m, p[(size_t)dh * W + dw]);
  out[idx] = m;
}

__global__ void maxpool_i32_bias(const int* __restrict__ y, const float* __restrict__ bias,
                                 float* __restrict__ out, int B, int Cch,
                                 int H, int W, int OH, int OW)
{
  long long total = (long long)B * Cch * OH * OW;
  long long idx = (long long)blockIdx.x * blockDim.x + threadIdx.x;
  if (idx >= total) return;
  int ow = (int)(idx % OW); long long t = idx / OW;
  int oh = (int)(t % OH); t /= OH;
  int c = (int)(t % Cch); int b = (int)(t / Cch);
  const size_t Ntot = (size_t)B * H * W;
  const int* p = y + (size_t)c * Ntot + (size_t)b * H * W + (size_t)(oh * 2) * W + ow * 2;
  int m = -2147483647;
  for (int dh = 0; dh < 3; ++dh)
    for (int dw = 0; dw < 3; ++dw)
      m = max(m, p[(size_t)dh * W + dw]);
  out[idx] = (float)m + bias[c];
}

__global__ void bn_stats_nchw(const float* __restrict__ x, const float* __restrict__ g,
                              const float* __restrict__ be, float* __restrict__ scale,
                              float* __restrict__ shift, int B, int C, int HW)
{
  const int c = blockIdx.x;
  const int total = B * HW;
  float s = 0.f, s2 = 0.f;
  for (int i = threadIdx.x; i < total; i += blockDim.x) {
    int b = i / HW, p = i - b * HW;
    float v = x[((size_t)b * C + c) * HW + p];
    s += v; s2 += v * v;
  }
  __shared__ float shs[256], shq[256];
  shs[threadIdx.x] = s; shq[threadIdx.x] = s2;
  __syncthreads();
  for (int st = 128; st > 0; st >>= 1) {
    if ((int)threadIdx.x < st) { shs[threadIdx.x] += shs[threadIdx.x + st];
                                 shq[threadIdx.x] += shq[threadIdx.x + st]; }
    __syncthreads();
  }
  if (threadIdx.x == 0) {
    float inv = 1.f / (float)total;
    float mean = shs[0] * inv;
    float var  = shq[0] * inv - mean * mean;
    float sc = g[c] * rsqrtf(var + BN_EPS);
    scale[c] = sc;
    shift[c] = be[c] - mean * sc;
  }
}

__global__ void bn_stats_gemm(const int* __restrict__ y, const float* __restrict__ bias,
                              const float* __restrict__ g, const float* __restrict__ be,
                              float* __restrict__ scale, float* __restrict__ shift, int Ntot)
{
  const int c = blockIdx.x;
  float s = 0.f, s2 = 0.f;
  for (int i = threadIdx.x; i < Ntot; i += blockDim.x) {
    float v = (float)y[(size_t)c * Ntot + i];
    s += v; s2 += v * v;
  }
  __shared__ float shs[256], shq[256];
  shs[threadIdx.x] = s; shq[threadIdx.x] = s2;
  __syncthreads();
  for (int st = 128; st > 0; st >>= 1) {
    if ((int)threadIdx.x < st) { shs[threadIdx.x] += shs[threadIdx.x + st];
                                 shq[threadIdx.x] += shq[threadIdx.x + st]; }
    __syncthreads();
  }
  if (threadIdx.x == 0) {
    float inv = 1.f / (float)Ntot;
    float mean = shs[0] * inv + bias[c];
    float var  = shq[0] * inv - (shs[0] * inv) * (shs[0] * inv);
    float sc = g[c] * rsqrtf(var + BN_EPS);
    scale[c] = sc;
    shift[c] = be[c] - mean * sc;
  }
}

__global__ void bn_sign_nchw(const float* __restrict__ x, const float* __restrict__ scale,
                             const float* __restrict__ shift, int8_t* __restrict__ s,
                             long long total, int C, int HW)
{
  long long idx = (long long)blockIdx.x * blockDim.x + threadIdx.x;
  if (idx >= total) return;
  int c = (int)((idx / HW) % C);
  float v = x[idx] * scale[c] + shift[c];
  s[idx] = (int8_t)(v >= 0.f ? 1 : -1);
}

__global__ void bn_sign_gemm(const int* __restrict__ y, const float* __restrict__ bias,
                             const float* __restrict__ scale, const float* __restrict__ shift,
                             int8_t* __restrict__ s, int B, int C, int HW)
{
  long long total = (long long)B * C * HW;
  long long idx = (long long)blockIdx.x * blockDim.x + threadIdx.x;
  if (idx >= total) return;
  int p = (int)(idx % HW); long long t = idx / HW;
  int c = (int)(t % C);    int b = (int)(t / C);
  float v = ((float)y[(size_t)c * ((size_t)B * HW) + (size_t)b * HW + p] + bias[c]) * scale[c] + shift[c];
  s[idx] = (int8_t)(v >= 0.f ? 1 : -1);
}

__global__ void bn_sign_gemm_T(const int* __restrict__ y, const float* __restrict__ bias,
                               const float* __restrict__ scale, const float* __restrict__ shift,
                               int8_t* __restrict__ s, int Mf, int Nb)
{
  long long total = (long long)Mf * Nb;
  long long idx = (long long)blockIdx.x * blockDim.x + threadIdx.x;
  if (idx >= total) return;
  int m = (int)(idx % Mf); int b = (int)(idx / Mf);
  float v = ((float)y[(size_t)m * Nb + b] + bias[m]) * scale[m] + shift[m];
  s[idx] = (int8_t)(v >= 0.f ? 1 : -1);
}

__global__ void im2col_i8(const int8_t* __restrict__ s, int8_t* __restrict__ Bm,
                          int B, int C, int H, int W, int OH, int OW, int KH, int pad)
{
  const int K = C * KH * KH;
  long long total = (long long)B * OH * OW * K;
  long long idx = (long long)blockIdx.x * blockDim.x + threadIdx.x;
  if (idx >= total) return;
  long long n = idx / K; int k = (int)(idx % K);
  int kw = k % KH; int kh = (k / KH) % KH; int c = k / (KH * KH);
  int ow = (int)(n % OW); int oh = (int)((n / OW) % OH); int b = (int)(n / ((long long)OW * OH));
  int ih = oh - pad + kh, iw = ow - pad + kw;
  int8_t v = 0;
  if ((unsigned)ih < (unsigned)H && (unsigned)iw < (unsigned)W)
    v = s[(((size_t)b * C + c) * H + ih) * W + iw];
  Bm[idx] = v;
}

__global__ void fc_bn_logsoftmax(const int* __restrict__ y, const float* __restrict__ bias,
                                 const float* __restrict__ scale, const float* __restrict__ shift,
                                 float* __restrict__ out, int Mf, int Nb)
{
  const int b = blockIdx.x;
  __shared__ float red[256];
  float mx = -3.402823466e38f;
  for (int j = threadIdx.x; j < Mf; j += blockDim.x) {
    float z = ((float)y[(size_t)j * Nb + b] + bias[j]) * scale[j] + shift[j];
    mx = fmaxf(mx, z);
  }
  red[threadIdx.x] = mx; __syncthreads();
  for (int st = 128; st > 0; st >>= 1) {
    if ((int)threadIdx.x < st) red[threadIdx.x] = fmaxf(red[threadIdx.x], red[threadIdx.x + st]);
    __syncthreads();
  }
  mx = red[0]; __syncthreads();
  float se = 0.f;
  for (int j = threadIdx.x; j < Mf; j += blockDim.x) {
    float z = ((float)y[(size_t)j * Nb + b] + bias[j]) * scale[j] + shift[j];
    se += expf(z - mx);
  }
  red[threadIdx.x] = se; __syncthreads();
  for (int st = 128; st > 0; st >>= 1) {
    if ((int)threadIdx.x < st) red[threadIdx.x] += red[threadIdx.x + st];
    __syncthreads();
  }
  float lse = logf(red[0]);
  for (int j = threadIdx.x; j < Mf; j += blockDim.x) {
    float z = ((float)y[(size_t)j * Nb + b] + bias[j]) * scale[j] + shift[j];
    out[(size_t)b * Mf + j] = z - mx - lse;
  }
}

static inline int blocks_for(long long total, int t) { return (int)((total + t - 1) / t); }

extern "C" void kernel_launch(void* const* d_in, const int* in_sizes, int n_in,
                              void* d_out, int out_size, void* d_ws, size_t ws_size,
                              hipStream_t stream)
{
  (void)in_sizes; (void)n_in; (void)out_size; (void)ws_size;
  const float* x   = (const float*)d_in[0];
  const float* w0  = (const float*)d_in[1];
  const float* b0  = (const float*)d_in[2];
  const float* g0  = (const float*)d_in[3];
  const float* be0 = (const float*)d_in[4];
  const float* w1  = (const float*)d_in[5];
  const float* b1  = (const float*)d_in[6];
  const float* g1  = (const float*)d_in[7];
  const float* be1 = (const float*)d_in[8];
  const float* w2  = (const float*)d_in[9];
  const float* b2  = (const float*)d_in[10];
  const float* g2  = (const float*)d_in[11];
  const float* be2 = (const float*)d_in[12];
  const float* w3  = (const float*)d_in[13];
  const float* b3  = (const float*)d_in[14];
  const float* g3  = (const float*)d_in[15];
  const float* be3 = (const float*)d_in[16];
  const float* w4  = (const float*)d_in[17];
  const float* b4  = (const float*)d_in[18];
  const float* g4  = (const float*)d_in[19];
  const float* be4 = (const float*)d_in[20];
  const float* fw0 = (const float*)d_in[21];
  const float* fb0 = (const float*)d_in[22];
  const float* fg0 = (const float*)d_in[23];
  const float* fbe0= (const float*)d_in[24];
  const float* fw1 = (const float*)d_in[25];
  const float* fb1 = (const float*)d_in[26];
  const float* fg1 = (const float*)d_in[27];
  const float* fbe1= (const float*)d_in[28];
  const float* fw2 = (const float*)d_in[29];
  const float* fb2 = (const float*)d_in[30];
  const float* fg2 = (const float*)d_in[31];
  const float* fbe2= (const float*)d_in[32];
  float* out = (float*)d_out;

  char* base = (char*)d_ws;
  size_t off = 0;
  auto alloc = [&](size_t bytes) -> char* {
    char* p = base + off;
    off = (off + bytes + 255) & ~(size_t)255;
    return p;
  };
  int8_t* w0s  = (int8_t*)alloc(64LL * 363);
  int8_t* w1s  = (int8_t*)alloc(192LL * 1600);
  int8_t* w2s  = (int8_t*)alloc(384LL * 1728);
  int8_t* w3s  = (int8_t*)alloc(384LL * 3456);
  int8_t* w4s  = (int8_t*)alloc(256LL * 3456);
  int8_t* fw0s = (int8_t*)alloc(4096LL * 9216);
  int8_t* fw1s = (int8_t*)alloc(4096LL * 4096);
  int8_t* fw2s = (int8_t*)alloc(1024LL * 4096);
  float* sc[8]; float* sh[8];
  for (int i = 0; i < 8; ++i) { sc[i] = (float*)alloc(4096 * 4); sh[i] = (float*)alloc(4096 * 4); }
  float* Y0  = (float*)alloc(128LL * 64 * 55 * 55 * 4);
  float* P0  = (float*)alloc(128LL * 64 * 27 * 27 * 4);
  int8_t* S1 = (int8_t*)alloc(128LL * 64 * 27 * 27);
  int8_t* B1 = (int8_t*)alloc(93312LL * 1600);
  int*    C1 = (int*)   alloc(192LL * 93312 * 4);
  float*  P1 = (float*) alloc(128LL * 192 * 13 * 13 * 4);
  int8_t* S2 = (int8_t*)alloc(128LL * 192 * 169);
  int8_t* B2 = (int8_t*)alloc(21632LL * 1728);
  int*    C2 = (int*)   alloc(384LL * 21632 * 4);
  int8_t* S3 = (int8_t*)alloc(128LL * 384 * 169);
  int8_t* B3 = (int8_t*)alloc(21632LL * 3456);
  int*    C3 = (int*)   alloc(384LL * 21632 * 4);
  int8_t* S4 = (int8_t*)alloc(128LL * 384 * 169);
  int8_t* B4 = (int8_t*)alloc(21632LL * 3456);
  int*    C4 = (int*)   alloc(256LL * 21632 * 4);
  float*  P4 = (float*) alloc(128LL * 256 * 36 * 4);
  int8_t* S5 = (int8_t*)alloc(128LL * 9216);
  int*   YF0 = (int*)   alloc(4096LL * 128 * 4);
  int8_t* H1 = (int8_t*)alloc(128LL * 4096);
  int*   YF1 = (int*)   alloc(4096LL * 128 * 4);
  int8_t* H2 = (int8_t*)alloc(128LL * 4096);
  int*   YF2 = (int*)   alloc(1024LL * 128 * 4);

  const int T = 256;
  pack_sign<<<blocks_for(64LL*363, T), T, 0, stream>>>(w0, w0s, 64LL*363, 64LL*363);
  pack_sign<<<blocks_for(192LL*1600, T), T, 0, stream>>>(w1, w1s, 192LL*1600, 192LL*1600);
  pack_sign<<<blocks_for(384LL*1728, T), T, 0, stream>>>(w2, w2s, 384LL*1728, 384LL*1728);
  pack_sign<<<blocks_for(384LL*3456, T), T, 0, stream>>>(w3, w3s, 384LL*3456, 384LL*3456);
  pack_sign<<<blocks_for(256LL*3456, T), T, 0, stream>>>(w4, w4s, 256LL*3456, 256LL*3456);
  pack_sign<<<blocks_for(4096LL*9216, T), T, 0, stream>>>(fw0, fw0s, 4096LL*9216, 4096LL*9216);
  pack_sign<<<blocks_for(4096LL*4096, T), T, 0, stream>>>(fw1, fw1s, 4096LL*4096, 4096LL*4096);
  pack_sign<<<blocks_for(1024LL*4096, T), T, 0, stream>>>(fw2, fw2s, 1000LL*4096, 1024LL*4096);

  conv0_direct<<<blocks_for(128LL*64*55*55, T), T, 0, stream>>>(x, w0s, b0, Y0);
  maxpool_f32<<<blocks_for(128LL*64*27*27, T), T, 0, stream>>>(Y0, P0, 128*64, 55, 55, 27, 27);
  bn_stats_nchw<<<64, T, 0, stream>>>(P0, g0, be0, sc[0], sh[0], 128, 64, 729);
  bn_sign_nchw<<<blocks_for(128LL*64*729, T), T, 0, stream>>>(P0, sc[0], sh[0], S1, 128LL*64*729, 64, 729);
  im2col_i8<<<blocks_for(93312LL*1600, T), T, 0, stream>>>(S1, B1, 128, 64, 27, 27, 27, 27, 5, 2);

  { long long tiles = 6LL * (93312/64); gemm_iu8_wmma<<<blocks_for(tiles*32, T), T, 0, stream>>>(w1s, B1, C1, 192, 93312, 1600, 6); }
  maxpool_i32_bias<<<blocks_for(128LL*192*169, T), T, 0, stream>>>(C1, b1, P1, 128, 192, 27, 27, 13, 13);
  bn_stats_nchw<<<192, T, 0, stream>>>(P1, g1, be1, sc[1], sh[1], 128, 192, 169);
  bn_sign_nchw<<<blocks_for(128LL*192*169, T), T, 0, stream>>>(P1, sc[1], sh[1], S2, 128LL*192*169, 192, 169);
  im2col_i8<<<blocks_for(21632LL*1728, T), T, 0, stream>>>(S2, B2, 128, 192, 13, 13, 13, 13, 3, 1);

  { long long tiles = 12LL * (21632/64); gemm_iu8_wmma<<<blocks_for(tiles*32, T), T, 0, stream>>>(w2s, B2, C2, 384, 21632, 1728, 12); }
  bn_stats_gemm<<<384, T, 0, stream>>>(C2, b2, g2, be2, sc[2], sh[2], 21632);
  bn_sign_gemm<<<blocks_for(128LL*384*169, T), T, 0, stream>>>(C2, b2, sc[2], sh[2], S3, 128, 384, 169);
  im2col_i8<<<blocks_for(21632LL*3456, T), T, 0, stream>>>(S3, B3, 128, 384, 13, 13, 13, 13, 3, 1);

  { long long tiles = 12LL * (21632/64); gemm_iu8_wmma<<<blocks_for(tiles*32, T), T, 0, stream>>>(w3s, B3, C3, 384, 21632, 3456, 12); }
  bn_stats_gemm<<<384, T, 0, stream>>>(C3, b3, g3, be3, sc[3], sh[3], 21632);
  bn_sign_gemm<<<blocks_for(128LL*384*169, T), T, 0, stream>>>(C3, b3, sc[3], sh[3], S4, 128, 384, 169);
  im2col_i8<<<blocks_for(21632LL*3456, T), T, 0, stream>>>(S4, B4, 128, 384, 13, 13, 13, 13, 3, 1);

  { long long tiles = 8LL * (21632/64); gemm_iu8_wmma<<<blocks_for(tiles*32, T), T, 0, stream>>>(w4s, B4, C4, 256, 21632, 3456, 8); }
  maxpool_i32_bias<<<blocks_for(128LL*256*36, T), T, 0, stream>>>(C4, b4, P4, 128, 256, 13, 13, 6, 6);
  bn_stats_nchw<<<256, T, 0, stream>>>(P4, g4, be4, sc[4], sh[4], 128, 256, 36);
  bn_sign_nchw<<<blocks_for(128LL*256*36, T), T, 0, stream>>>(P4, sc[4], sh[4], S5, 128LL*256*36, 256, 36);

  { long long tiles = 128LL * 2; gemm_iu8_wmma<<<blocks_for(tiles*32, T), T, 0, stream>>>(fw0s, S5, YF0, 4096, 128, 9216, 128); }
  bn_stats_gemm<<<4096, T, 0, stream>>>(YF0, fb0, fg0, fbe0, sc[5], sh[5], 128);
  bn_sign_gemm_T<<<blocks_for(4096LL*128, T), T, 0, stream>>>(YF0, fb0, sc[5], sh[5], H1, 4096, 128);

  { long long tiles = 128LL * 2; gemm_iu8_wmma<<<blocks_for(tiles*32, T), T, 0, stream>>>(fw1s, H1, YF1, 4096, 128, 4096, 128); }
  bn_stats_gemm<<<4096, T, 0, stream>>>(YF1, fb1, fg1, fbe1, sc[6], sh[6], 128);
  bn_sign_gemm_T<<<blocks_for(4096LL*128, T), T, 0, stream>>>(YF1, fb1, sc[6], sh[6], H2, 4096, 128);

  { long long tiles = 32LL * 2; gemm_iu8_wmma<<<blocks_for(tiles*32, T), T, 0, stream>>>(fw2s, H2, YF2, 1000, 128, 4096, 32); }
  bn_stats_gemm<<<1000, T, 0, stream>>>(YF2, fb2, fg2, fbe2, sc[7], sh[7], 128);
  fc_bn_logsoftmax<<<128, T, 0, stream>>>(YF2, fb2, sc[7], sh[7], out, 1000, 128);
}
